// Decoder_60868276519063
// MI455X (gfx1250) — matrix operand order repaired and checked by reading, not compiled, not run on hardware
//
#include <hip/hip_runtime.h>
#include <hip/hip_bf16.h>
#include <math.h>

// Problem constants (from reference)
#define BB   32
#define TT   800
#define EE   512
#define DD   1024
#define VV   5000
#define VPAD 5008          // V padded to multiple of 16 for WMMA N tiles
#define LLEN 149
#define SSTEPS 150         // L+1
#define AA   512
#define KC0  2560          // D + E + D  (concat [ey; att_c; z0] for LSTM0 fused GEMM)
#define KC1  2048          // D + D      (concat [z0; z1] for LSTM1 fused GEMM)
#define SOS_TOK 1
#define EOS_TOK 2

typedef __bf16 bf16_t;
typedef __attribute__((ext_vector_type(16))) __bf16 v16bf;
typedef __attribute__((ext_vector_type(8)))  __bf16 v8bf;
typedef __attribute__((ext_vector_type(8)))  float  v8f;

__device__ __forceinline__ bf16_t f2bf(float f) {
    union { float f; unsigned int u; } v; v.f = f;
    unsigned int r = v.u + 0x7FFFu + ((v.u >> 16) & 1u);   // round-to-nearest-even
    union { unsigned short s; bf16_t b; } o;
    o.s = (unsigned short)(r >> 16);
    return o.b;
}
__device__ __forceinline__ float bf2f(bf16_t b) {
    union { unsigned short s; bf16_t b; } u; u.b = b;
    union { unsigned int i; float f; } v; v.i = ((unsigned int)u.s) << 16;
    return v.f;
}
__device__ __forceinline__ float sigmf(float x) { return 1.0f / (1.0f + expf(-x)); }

// ---------------------------------------------------------------------------
// WMMA GEMM:  C[M,N] = A[M,K](bf16,row major,lda) x Bt[K,N](bf16,row major,ldb)
//             (+ bias[n]).  Result stored fp32 (C) or bf16 (Cbf) per call.
// Fragment layouts per CDNA5 ISA 7.12.2:
//   A (16x32 MxK): lanes 0-15 rows M=0..15 hold K={0..7,16..23};
//                  lanes 16-31 same rows hold K={8..15,24..31}
//   B (32x16 KxN): lane l holds column N=l%16, same K order as A
// block = 256 threads = 8 waves; wave owns one 16-wide N tile and both 16-row
// M tiles of a 32-row M block (grid.y). M,K multiples of 32; N multiple of 16.
// Branchless next-tile prefetch: over-run stays inside the workspace because
// every weight buffer is followed by other live workspace buffers.
// ---------------------------------------------------------------------------
__global__ void k_gemm_bf16(const bf16_t* __restrict__ A, int lda,
                            const bf16_t* __restrict__ Bt, int ldb,
                            float* __restrict__ C, bf16_t* __restrict__ Cbf, int ldc,
                            const float* __restrict__ bias,
                            int N, int K) {
    const int lane = threadIdx.x & 31;
    const int wave = threadIdx.x >> 5;
    const int n0 = (blockIdx.x * (blockDim.x >> 5) + wave) * 16;
    const int m0 = blockIdx.y * 32;
    if (n0 >= N) return;

    const size_t bstep = (size_t)32 * ldb;
    const bf16_t* pB  = Bt + (size_t)((lane >> 4) << 3) * ldb + n0 + (lane & 15);
    const bf16_t* pA0 = A + (size_t)(m0 + (lane & 15)) * lda + ((lane >> 4) << 3);
    const bf16_t* pA1 = pA0 + (size_t)16 * lda;

    v8f acc0 = {};
    v8f acc1 = {};
    for (int k0 = 0; k0 < K; k0 += 32) {
        __builtin_prefetch(pB + bstep, 0, 1);       // global_prefetch, branchless

        v8bf a0lo = *(const v8bf*)(pA0);
        v8bf a0hi = *(const v8bf*)(pA0 + 16);
        v8bf a1lo = *(const v8bf*)(pA1);
        v8bf a1hi = *(const v8bf*)(pA1 + 16);
        v16bf bfr, a0, a1;
#pragma unroll
        for (int i = 0; i < 8; ++i) {
            bfr[i] = pB[(size_t)i * ldb]; bfr[i + 8] = pB[(size_t)(16 + i) * ldb];
            a0[i] = a0lo[i]; a0[i + 8] = a0hi[i];
            a1[i] = a1lo[i]; a1[i + 8] = a1hi[i];
        }
        acc0 = __builtin_amdgcn_wmma_f32_16x16x32_bf16(false, a0, false, bfr,
                                                       (short)0, acc0, false, false);
        acc1 = __builtin_amdgcn_wmma_f32_16x16x32_bf16(false, a1, false, bfr,
                                                       (short)0, acc1, false, false);
        pB  += bstep;
        pA0 += 32;
        pA1 += 32;
    }

    const int n  = n0 + (lane & 15);
    const int mo = (lane >> 4) << 3;           // 0 (lanes 0-15) or 8 (lanes 16-31)
    const float bv = bias ? bias[n] : 0.0f;
    if (Cbf) {
#pragma unroll
        for (int j = 0; j < 8; ++j)
            Cbf[(size_t)(m0 + mo + j) * ldc + n] = f2bf(acc0[j] + bv);
#pragma unroll
        for (int j = 0; j < 8; ++j)
            Cbf[(size_t)(m0 + 16 + mo + j) * ldc + n] = f2bf(acc1[j] + bv);
    } else {
#pragma unroll
        for (int j = 0; j < 8; ++j)
            C[(size_t)(m0 + mo + j) * ldc + n] = acc0[j] + bv;
#pragma unroll
        for (int j = 0; j < 8; ++j)
            C[(size_t)(m0 + 16 + mo + j) * ldc + n] = acc1[j] + bv;
    }
}

// ---------------------------------------------------------------------------
// One-time preparation kernels
// ---------------------------------------------------------------------------
__global__ void k_zero_f(float* p, int n) {
    int i = blockIdx.x * blockDim.x + threadIdx.x;
    if (i < n) p[i] = 0.0f;
}
__global__ void k_zero_bf(bf16_t* p, int n) {
    int i = blockIdx.x * blockDim.x + threadIdx.x;
    if (i < n) p[i] = f2bf(0.0f);
}
__global__ void k_cvt_bf16(const float* __restrict__ s, bf16_t* __restrict__ d, int n) {
    int i = blockIdx.x * blockDim.x + threadIdx.x;
    if (i < n) d[i] = f2bf(s[i]);
}
// Wcat0_t[k, n] : k<1536 -> W_ih0[n, k] ; else W_hh0[n, k-1536]   (K=2560, N=4096)
__global__ void k_build_w0(const float* __restrict__ Wih, const float* __restrict__ Whh,
                           bf16_t* __restrict__ dst) {
    int idx = blockIdx.x * blockDim.x + threadIdx.x;
    if (idx >= KC0 * 4 * DD) return;
    int k = idx / (4 * DD), n = idx % (4 * DD);
    float v = (k < (DD + EE)) ? Wih[(size_t)n * (DD + EE) + k]
                              : Whh[(size_t)n * DD + (k - (DD + EE))];
    dst[idx] = f2bf(v);
}
// Wcat1_t[k, n] : k<1024 -> W_ih1[n, k] ; else W_hh1[n, k-1024]   (K=2048, N=4096)
__global__ void k_build_w1(const float* __restrict__ Wih, const float* __restrict__ Whh,
                           bf16_t* __restrict__ dst) {
    int idx = blockIdx.x * blockDim.x + threadIdx.x;
    if (idx >= KC1 * 4 * DD) return;
    int k = idx / (4 * DD), n = idx % (4 * DD);
    float v = (k < DD) ? Wih[(size_t)n * DD + k] : Whh[(size_t)n * DD + (k - DD)];
    dst[idx] = f2bf(v);
}
// Wout_t[k, n] = n<V ? W_out[n, k] : 0     (K=1024, N=5008)
__global__ void k_build_wout(const float* __restrict__ Wout, bf16_t* __restrict__ dst) {
    int idx = blockIdx.x * blockDim.x + threadIdx.x;
    if (idx >= DD * VPAD) return;
    int k = idx / VPAD, n = idx % VPAD;
    dst[idx] = f2bf((n < VV) ? Wout[(size_t)n * DD + k] : 0.0f);
}
__global__ void k_bias_sum(const float* a, const float* b, float* d, int n) {
    int i = blockIdx.x * blockDim.x + threadIdx.x;
    if (i < n) d[i] = a[i] + b[i];
}
__global__ void k_bias_out(const float* b, float* d) {
    int i = blockIdx.x * blockDim.x + threadIdx.x;
    if (i < VPAD) d[i] = (i < VV) ? b[i] : -1e30f;   // pad columns never win
}
// eys_bf[b, s, :] = bf16(embed[ys_in[b,s]])
__global__ void k_embed(const float* __restrict__ embed, const int* __restrict__ ys,
                        bf16_t* __restrict__ eys) {
    int bs = blockIdx.x;
    int b = bs / SSTEPS, s = bs % SSTEPS;
    int tok = (s == 0) ? SOS_TOK : ys[b * LLEN + (s - 1)];
    const float* src = embed + (size_t)tok * DD;
    bf16_t* dst = eys + (size_t)bs * DD;
    for (int j = threadIdx.x; j < DD; j += blockDim.x) dst[j] = f2bf(src[j]);
}

// ---------------------------------------------------------------------------
// Per-step kernels
// ---------------------------------------------------------------------------
__global__ void k_pack_ey(const bf16_t* __restrict__ eys, int s, bf16_t* __restrict__ xcat0) {
    int i = blockIdx.x * blockDim.x + threadIdx.x;
    if (i >= BB * DD) return;
    int b = i / DD, j = i % DD;
    xcat0[(size_t)b * KC0 + j] = eys[((size_t)b * SSTEPS + s) * DD + j];
}

// e[b,t] = sum_a tanh(pe_bf[b,t,a] + dproj[b,a]) * v_att[a]   (wave per row)
__global__ void k_energy(const bf16_t* __restrict__ pe, const float* __restrict__ dproj,
                         const float* __restrict__ v_att, const int* __restrict__ hlen,
                         float* __restrict__ ebuf) {
    int lane = threadIdx.x & 31;
    int wave = threadIdx.x >> 5;
    int b = blockIdx.y;
    int t = blockIdx.x * (blockDim.x >> 5) + wave;
    if (t >= TT) return;
    const bf16_t* pr = pe + ((size_t)b * TT + t) * AA;
    const float*  dp = dproj + (size_t)b * AA;
    float s = 0.0f;
#pragma unroll
    for (int p = 0; p < AA; p += 256) {            // 2 passes, 8 elems/lane each
        int a = p + lane * 8;
        v8bf pv = *(const v8bf*)(pr + a);
#pragma unroll
        for (int i = 0; i < 8; ++i)
            s += tanhf(bf2f(pv[i]) + dp[a + i]) * v_att[a + i];
    }
#pragma unroll
    for (int off = 16; off; off >>= 1) s += __shfl_down(s, off, 32);
    if (lane == 0) ebuf[b * TT + t] = (t < hlen[b]) ? s : -1e30f;
}

__global__ void k_softmax(const float* __restrict__ ebuf, float* __restrict__ wbuf) {
    __shared__ float red[1024];
    int b = blockIdx.x, tid = threadIdx.x;
    float m = -1e30f;
    for (int t = tid; t < TT; t += blockDim.x) m = fmaxf(m, ebuf[b * TT + t]);
    red[tid] = m; __syncthreads();
    for (int s = blockDim.x / 2; s; s >>= 1) {
        if (tid < s) red[tid] = fmaxf(red[tid], red[tid + s]);
        __syncthreads();
    }
    m = red[0]; __syncthreads();
    float sum = 0.0f;
    for (int t = tid; t < TT; t += blockDim.x) sum += expf(ebuf[b * TT + t] - m);
    red[tid] = sum; __syncthreads();
    for (int s = blockDim.x / 2; s; s >>= 1) {
        if (tid < s) red[tid] += red[tid + s];
        __syncthreads();
    }
    float inv = 1.0f / red[0];
    for (int t = tid; t < TT; t += blockDim.x)
        wbuf[b * TT + t] = expf(ebuf[b * TT + t] - m) * inv;
}

// att_c[b,e] = sum_t w[b,t] * hpad_bf[b,t,e]  -> bf16 into xcat0 att slot
__global__ void k_attc(const bf16_t* __restrict__ hpad_bf, const float* __restrict__ wbuf,
                       const int* __restrict__ hlen, bf16_t* __restrict__ xcat0) {
    int b = blockIdx.x, e = threadIdx.x;    // blockDim == EE
    int len = hlen[b];
    const bf16_t* hb = hpad_bf + (size_t)b * TT * EE + e;
    const float*  wb = wbuf + (size_t)b * TT;
    float s = 0.0f;
    for (int t = 0; t < len; ++t) s += wb[t] * bf2f(hb[(size_t)t * EE]);
    xcat0[(size_t)b * KC0 + DD + e] = f2bf(s);
}

// LSTM cell pointwise (gate order i,f,g,o). Writes h (bf16) into up to 2 slots.
__global__ void k_lstm_cell(const float* __restrict__ g, float* __restrict__ c,
                            bf16_t* __restrict__ h0, int stride0, int off0,
                            bf16_t* __restrict__ h1, int stride1, int off1) {
    int i = blockIdx.x * blockDim.x + threadIdx.x;
    if (i >= BB * DD) return;
    int b = i / DD, j = i % DD;
    const float* gb = g + (size_t)b * 4 * DD;
    float ig = sigmf(gb[j]);
    float fg = sigmf(gb[DD + j]);
    float gg = tanhf(gb[2 * DD + j]);
    float og = sigmf(gb[3 * DD + j]);
    float cn = fg * c[i] + ig * gg;
    c[i] = cn;
    bf16_t h = f2bf(og * tanhf(cn));
    h0[(size_t)b * stride0 + off0 + j] = h;
    if (h1) h1[(size_t)b * stride1 + off1 + j] = h;
}

// NLL + argmax accumulation over one step's logits [BB, VPAD]
__global__ void k_loss(const float* __restrict__ logits, const int* __restrict__ ys,
                       int s, float* __restrict__ accum) {
    __shared__ float red[1024];
    __shared__ int  redi[1024];
    int b = blockIdx.x, tid = threadIdx.x;
    const float* lr = logits + (size_t)b * VPAD;
    int y = (s < LLEN) ? ys[b * LLEN + s] : EOS_TOK;
    float m = -1e30f; int mi = 0;
    for (int n = tid; n < VPAD; n += blockDim.x) {
        float v = lr[n];
        if (v > m) { m = v; mi = n; }
    }
    red[tid] = m; redi[tid] = mi; __syncthreads();
    for (int st = blockDim.x / 2; st; st >>= 1) {
        if (tid < st && red[tid + st] > red[tid]) { red[tid] = red[tid + st]; redi[tid] = redi[tid + st]; }
        __syncthreads();
    }
    float mx = red[0]; int am = redi[0]; __syncthreads();
    float sum = 0.0f;
    for (int n = tid; n < VPAD; n += blockDim.x) sum += expf(lr[n] - mx);
    red[tid] = sum; __syncthreads();
    for (int st = blockDim.x / 2; st; st >>= 1) {
        if (tid < st) red[tid] += red[tid + st];
        __syncthreads();
    }
    if (tid == 0) {
        float nll = (mx + logf(red[0])) - lr[y];
        atomicAdd(&accum[0], nll);
        atomicAdd(&accum[1], (am == y) ? 1.0f : 0.0f);
    }
}

__global__ void k_finalize(const float* __restrict__ accum, float* __restrict__ out) {
    if (threadIdx.x == 0 && blockIdx.x == 0) {
        float denom = (float)(BB * SSTEPS);
        out[0] = accum[0] / denom * (float)LLEN;
        out[1] = accum[1] / denom;
    }
}

// ---------------------------------------------------------------------------
extern "C" void kernel_launch(void* const* d_in, const int* in_sizes, int n_in,
                              void* d_out, int out_size, void* d_ws, size_t ws_size,
                              hipStream_t stream) {
    (void)in_sizes; (void)n_in; (void)out_size; (void)ws_size;
    const float* hpad  = (const float*)d_in[0];
    const int*   hlen  = (const int*)  d_in[1];
    const int*   ys    = (const int*)  d_in[2];
    const float* embed = (const float*)d_in[3];
    const float* W_out = (const float*)d_in[4];
    const float* b_out = (const float*)d_in[5];
    const float* W_enc = (const float*)d_in[6];
    const float* W_dec = (const float*)d_in[7];
    const float* v_att = (const float*)d_in[8];
    const float* W_ih0 = (const float*)d_in[9];
    const float* b_ih0 = (const float*)d_in[10];
    const float* W_hh0 = (const float*)d_in[11];
    const float* b_hh0 = (const float*)d_in[12];
    const float* W_ih1 = (const float*)d_in[13];
    const float* b_ih1 = (const float*)d_in[14];
    const float* W_hh1 = (const float*)d_in[15];
    const float* b_hh1 = (const float*)d_in[16];
    float* out = (float*)d_out;

    // ---- workspace carve-up (256B aligned) ----
    char* ws = (char*)d_ws;
    size_t off = 0;
    auto alloc = [&](size_t bytes) -> char* {
        char* p = ws + off;
        off = (off + bytes + 255) & ~(size_t)255;
        return p;
    };
    bf16_t* hpad_bf = (bf16_t*)alloc((size_t)BB * TT * EE * 2);
    bf16_t* pe_bf   = (bf16_t*)alloc((size_t)BB * TT * AA * 2);
    bf16_t* eys     = (bf16_t*)alloc((size_t)BB * SSTEPS * DD * 2);
    bf16_t* Wenc_bf = (bf16_t*)alloc((size_t)EE * AA * 2);
    bf16_t* Wdec_bf = (bf16_t*)alloc((size_t)DD * AA * 2);
    bf16_t* W0t     = (bf16_t*)alloc((size_t)KC0 * 4 * DD * 2);
    bf16_t* W1t     = (bf16_t*)alloc((size_t)KC1 * 4 * DD * 2);
    bf16_t* Wot     = (bf16_t*)alloc((size_t)DD * VPAD * 2);
    float*  bias0   = (float*) alloc(4 * DD * 4);
    float*  bias1   = (float*) alloc(4 * DD * 4);
    float*  biaso   = (float*) alloc(VPAD * 4);
    bf16_t* xcat0   = (bf16_t*)alloc((size_t)BB * KC0 * 2);
    bf16_t* xcat1   = (bf16_t*)alloc((size_t)BB * KC1 * 2);
    float*  dproj   = (float*) alloc((size_t)BB * AA * 4);
    float*  ebuf    = (float*) alloc((size_t)BB * TT * 4);
    float*  wbuf    = (float*) alloc((size_t)BB * TT * 4);
    float*  g0      = (float*) alloc((size_t)BB * 4 * DD * 4);
    float*  g1      = (float*) alloc((size_t)BB * 4 * DD * 4);
    float*  c0      = (float*) alloc((size_t)BB * DD * 4);
    float*  c1      = (float*) alloc((size_t)BB * DD * 4);
    float*  logits  = (float*) alloc((size_t)BB * VPAD * 4);
    float*  accum   = (float*) alloc(2 * 4);

    auto cdiv = [](int a, int b) { return (a + b - 1) / b; };

    // ---- init (must run every call: deterministic state) ----
    k_zero_f <<<cdiv(BB * DD, 256), 256, 0, stream>>>(c0, BB * DD);
    k_zero_f <<<cdiv(BB * DD, 256), 256, 0, stream>>>(c1, BB * DD);
    k_zero_bf<<<cdiv(BB * KC0, 256), 256, 0, stream>>>(xcat0, BB * KC0);
    k_zero_bf<<<cdiv(BB * KC1, 256), 256, 0, stream>>>(xcat1, BB * KC1);
    k_zero_f <<<1, 32, 0, stream>>>(accum, 2);

    // ---- one-time precompute ----
    k_cvt_bf16<<<cdiv(BB * TT * EE, 256), 256, 0, stream>>>(hpad, hpad_bf, BB * TT * EE);
    k_cvt_bf16<<<cdiv(EE * AA, 256), 256, 0, stream>>>(W_enc, Wenc_bf, EE * AA);
    k_cvt_bf16<<<cdiv(DD * AA, 256), 256, 0, stream>>>(W_dec, Wdec_bf, DD * AA);
    k_build_w0  <<<cdiv(KC0 * 4 * DD, 256), 256, 0, stream>>>(W_ih0, W_hh0, W0t);
    k_build_w1  <<<cdiv(KC1 * 4 * DD, 256), 256, 0, stream>>>(W_ih1, W_hh1, W1t);
    k_build_wout<<<cdiv(DD * VPAD, 256), 256, 0, stream>>>(W_out, Wot);
    k_bias_sum<<<cdiv(4 * DD, 256), 256, 0, stream>>>(b_ih0, b_hh0, bias0, 4 * DD);
    k_bias_sum<<<cdiv(4 * DD, 256), 256, 0, stream>>>(b_ih1, b_hh1, bias1, 4 * DD);
    k_bias_out<<<cdiv(VPAD, 256), 256, 0, stream>>>(b_out, biaso);
    k_embed<<<BB * SSTEPS, 256, 0, stream>>>(embed, ys, eys);

    // pe = hpad @ W_enc  : M = B*T = 25600, K = E = 512, N = A = 512, bf16 out
    k_gemm_bf16<<<dim3(cdiv(AA, 128), (BB * TT) / 32), 256, 0, stream>>>(
        hpad_bf, EE, Wenc_bf, AA, nullptr, pe_bf, AA, nullptr, AA, EE);

    // ---- sequential decode: 150 steps ----
    for (int s = 0; s < SSTEPS; ++s) {
        // stage embedding for this step into xcat0[:, 0:D]
        k_pack_ey<<<cdiv(BB * DD, 256), 256, 0, stream>>>(eys, s, xcat0);

        // dproj = z0 @ W_dec   (A = xcat0 z0-slot, lda = KC0)
        k_gemm_bf16<<<dim3(cdiv(AA, 128), 1), 256, 0, stream>>>(
            xcat0 + (DD + EE), KC0, Wdec_bf, AA, dproj, nullptr, AA, nullptr, AA, DD);

        // attention energies, softmax, context
        k_energy<<<dim3(cdiv(TT, 8), BB), 256, 0, stream>>>(pe_bf, dproj, v_att, hlen, ebuf);
        k_softmax<<<BB, 1024, 0, stream>>>(ebuf, wbuf);
        k_attc<<<BB, EE, 0, stream>>>(hpad_bf, wbuf, hlen, xcat0);

        // LSTM0 gates: [ey; att_c; z0] @ Wcat0_t + bias0
        k_gemm_bf16<<<dim3(cdiv(4 * DD, 128), 1), 256, 0, stream>>>(
            xcat0, KC0, W0t, 4 * DD, g0, nullptr, 4 * DD, bias0, 4 * DD, KC0);
        k_lstm_cell<<<cdiv(BB * DD, 256), 256, 0, stream>>>(
            g0, c0, xcat0, KC0, DD + EE, xcat1, KC1, 0);

        // LSTM1 gates: [z0; z1] @ Wcat1_t + bias1
        k_gemm_bf16<<<dim3(cdiv(4 * DD, 128), 1), 256, 0, stream>>>(
            xcat1, KC1, W1t, 4 * DD, g1, nullptr, 4 * DD, bias1, 4 * DD, KC1);
        k_lstm_cell<<<cdiv(BB * DD, 256), 256, 0, stream>>>(
            g1, c1, xcat1, KC1, DD, nullptr, 0, 0);

        // logits = z1 @ W_out^T + b_out (N padded to 5008)
        k_gemm_bf16<<<dim3(cdiv(VPAD, 128), 1), 256, 0, stream>>>(
            xcat1 + DD, KC1, Wot, VPAD, logits, nullptr, VPAD, biaso, VPAD, DD);

        // cross-entropy + accuracy accumulation
        k_loss<<<BB, 1024, 0, stream>>>(logits, ys, s, accum);
    }

    k_finalize<<<1, 32, 0, stream>>>(accum, out);
}
